// AttentionBlock_6012954214619
// MI455X (gfx1250) — matrix operand order repaired and checked by reading, not compiled, not run on hardware
//
#include <hip/hip_runtime.h>
#include <hip/hip_bf16.h>

typedef __bf16 bf16_t;
typedef __attribute__((ext_vector_type(16))) __bf16 v16bf;
typedef __attribute__((ext_vector_type(8)))  float  v8f;

union Frag32 { v16bf v; uint4 q[2]; };

#define B_ 16
#define C_ 512
#define N_ 1024

// ---------------------------------------------------------------- weights f32 -> bf16
__global__ __launch_bounds__(256) void cvt_f32_bf16(const float* __restrict__ in,
                                                    bf16_t* __restrict__ out, int n) {
  int i = blockIdx.x * 256 + threadIdx.x;
  if (i < n) out[i] = (bf16_t)in[i];
}

// ---------------------------------------------------------------- GroupNorm -> Ht[b][pix][C] bf16
__global__ __launch_bounds__(256) void groupnorm_to_bf16(
    const float* __restrict__ x, const float* __restrict__ scale,
    const float* __restrict__ bias, bf16_t* __restrict__ Ht) {
  int b = blockIdx.x >> 5;
  int g = blockIdx.x & 31;
  int cbase = g * 16;
  const float* xp = x + ((size_t)b * C_ + cbase) * N_;   // 16 ch x 1024 pix contiguous
  int tid = threadIdx.x;
  float s = 0.f, s2 = 0.f;
  for (int i = tid; i < 16 * N_; i += 256) { float v = xp[i]; s += v; s2 += v * v; }
  __shared__ float rs[256], rq[256];
  rs[tid] = s; rq[tid] = s2;
  __syncthreads();
  for (int st = 128; st > 0; st >>= 1) {
    if (tid < st) { rs[tid] += rs[tid + st]; rq[tid] += rq[tid + st]; }
    __syncthreads();
  }
  float mean = rs[0] * (1.f / 16384.f);
  float var  = rq[0] * (1.f / 16384.f) - mean * mean;
  float rstd = rsqrtf(var + 1e-5f);
  for (int i = tid; i < 16 * N_; i += 256) {
    int c = cbase + (i >> 10);
    int p = i & 1023;
    float v = (xp[i] - mean) * rstd * scale[c] + bias[c];
    Ht[((size_t)b * N_ + p) * C_ + c] = (bf16_t)v;   // transposed store
  }
}

// ---------------------------------------------------------------- generic C = A(MxK) @ B(NxK)^T
// 64x64 C macro-tile per wave: 4x4 fragments, 16 wmma per 16 b128 loads per k-step.
// mode 0: bf16 out, bias per column   (Q/K:  Ht @ Wq^T -> [pix][C])
// mode 1: bf16 out, bias per row      (V:    Wv @ Ht^T -> [C][pix])
// mode 2: f32  out, bias per row + x  (proj: Wp @ Ot^T -> [C][pix] + residual)
__global__ __launch_bounds__(128) void gemm_nt_bf16(
    const bf16_t* __restrict__ A, const bf16_t* __restrict__ Bm,
    const float* __restrict__ bias, const float* __restrict__ residual,
    void* __restrict__ Cout,
    int M, int N, int Kd,
    long long sA, long long sB, long long sC, int mode) {
  int batch = blockIdx.z;
  int tile  = blockIdx.x * blockDim.y + threadIdx.y;   // 4 waves / block
  int tiles_n = N >> 6;                                 // N / 64
  int tm = tile / tiles_n, tn = tile % tiles_n;
  int lane = threadIdx.x;
  int half = lane >> 4, l16 = lane & 15;

  const bf16_t* Ab = A  + (size_t)batch * sA;
  const bf16_t* Bb = Bm + (size_t)batch * sB;
  const bf16_t* aB[4];
  const bf16_t* bB[4];
#pragma unroll
  for (int i = 0; i < 4; ++i) {
    aB[i] = Ab + ((size_t)(tm * 64 + i * 16 + l16)) * Kd + half * 8;
    bB[i] = Bb + ((size_t)(tn * 64 + i * 16 + l16)) * Kd + half * 8;
  }

  v8f acc[4][4] = {};
  for (int kk = 0; kk < Kd; kk += 32) {
    Frag32 a[4], b[4];
#pragma unroll
    for (int i = 0; i < 4; ++i) {
      a[i].q[0] = *(const uint4*)(aB[i] + kk);
      a[i].q[1] = *(const uint4*)(aB[i] + kk + 16);
      b[i].q[0] = *(const uint4*)(bB[i] + kk);
      b[i].q[1] = *(const uint4*)(bB[i] + kk + 16);
    }
#pragma unroll
    for (int mi = 0; mi < 4; ++mi)
#pragma unroll
      for (int ni = 0; ni < 4; ++ni)
        acc[mi][ni] = __builtin_amdgcn_wmma_f32_16x16x32_bf16(
            false, a[mi].v, false, b[ni].v, (short)0, acc[mi][ni], false, false);
  }

  if (mode == 0) {
    bf16_t* C = (bf16_t*)Cout + (size_t)batch * sC;
#pragma unroll
    for (int ni = 0; ni < 4; ++ni) {
      int col = tn * 64 + ni * 16 + l16;
      float bv = bias[col];
#pragma unroll
      for (int mi = 0; mi < 4; ++mi)
#pragma unroll
        for (int r = 0; r < 8; ++r) {
          int row = tm * 64 + mi * 16 + r + half * 8;
          C[(size_t)row * N + col] = (bf16_t)(acc[mi][ni][r] + bv);
        }
    }
  } else if (mode == 1) {
    bf16_t* C = (bf16_t*)Cout + (size_t)batch * sC;
#pragma unroll
    for (int ni = 0; ni < 4; ++ni) {
      int col = tn * 64 + ni * 16 + l16;
#pragma unroll
      for (int mi = 0; mi < 4; ++mi)
#pragma unroll
        for (int r = 0; r < 8; ++r) {
          int row = tm * 64 + mi * 16 + r + half * 8;
          C[(size_t)row * N + col] = (bf16_t)(acc[mi][ni][r] + bias[row]);
        }
    }
  } else {
    float* C = (float*)Cout + (size_t)batch * sC;
    const float* R = residual + (size_t)batch * sC;
#pragma unroll
    for (int ni = 0; ni < 4; ++ni) {
      int col = tn * 64 + ni * 16 + l16;
#pragma unroll
      for (int mi = 0; mi < 4; ++mi)
#pragma unroll
        for (int r = 0; r < 8; ++r) {
          int row = tm * 64 + mi * 16 + r + half * 8;
          size_t idx = (size_t)row * N + col;
          C[idx] = R[idx] + bias[row] + acc[mi][ni][r];
        }
    }
  }
}

// ---------------------------------------------------------------- fused attention per (b, head, 16 queries)
__global__ __launch_bounds__(32) void attn_fused(
    const bf16_t* __restrict__ Qt, const bf16_t* __restrict__ Kt,
    const bf16_t* __restrict__ V, bf16_t* __restrict__ Ot) {
  __shared__ float sS[16 * N_];   // 64 KB: f32 scores, later aliased by bf16 probs (in-place)
  int b = blockIdx.z, h = blockIdx.y, qb = blockIdx.x;
  int lane = threadIdx.x;
  int half = lane >> 4, l16 = lane & 15;

  const bf16_t* Qb = Qt + ((size_t)b * N_ + qb * 16) * C_ + h * 64;
  const bf16_t* Kb = Kt + (size_t)b * N_ * C_ + h * 64;
  const bf16_t* Vb = V  + ((size_t)b * C_ + h * 64) * N_;

  // phase 1: S = (Q^T K) / sqrt(d), 64 tiles of 16x16, K=64.
  // Q fragments are invariant across key tiles -> load once, keep in VGPRs.
  const bf16_t* aq = Qb + (size_t)l16 * C_ + half * 8;
  Frag32 a0, a1;
  a0.q[0] = *(const uint4*)(aq);
  a0.q[1] = *(const uint4*)(aq + 16);
  a1.q[0] = *(const uint4*)(aq + 32);
  a1.q[1] = *(const uint4*)(aq + 48);

  for (int mt = 0; mt < 64; ++mt) {
    const bf16_t* bk = Kb + (size_t)(mt * 16 + l16) * C_ + half * 8;
    Frag32 b0, b1;
    b0.q[0] = *(const uint4*)(bk);
    b0.q[1] = *(const uint4*)(bk + 16);
    b1.q[0] = *(const uint4*)(bk + 32);
    b1.q[1] = *(const uint4*)(bk + 48);
    v8f acc = {};
    acc = __builtin_amdgcn_wmma_f32_16x16x32_bf16(false, a0.v, false, b0.v,
                                                  (short)0, acc, false, false);
    acc = __builtin_amdgcn_wmma_f32_16x16x32_bf16(false, a1.v, false, b1.v,
                                                  (short)0, acc, false, false);
#pragma unroll
    for (int r = 0; r < 8; ++r)
      sS[(r + half * 8) * N_ + mt * 16 + l16] = acc[r] * 0.125f;
  }
  __syncthreads();

  // phase 2: row softmax; write normalized probs bf16 in-place (write byte 2j trails read byte 4j)
  for (int r = 0; r < 16; ++r) {
    float* rp = &sS[r * N_];
    float mx = -3.4e38f;
    for (int j = lane; j < N_; j += 32) mx = fmaxf(mx, rp[j]);
    for (int off = 16; off > 0; off >>= 1) mx = fmaxf(mx, __shfl_xor(mx, off, 32));
    float sum = 0.f;
    for (int j = lane; j < N_; j += 32) sum += __expf(rp[j] - mx);
    for (int off = 16; off > 0; off >>= 1) sum += __shfl_xor(sum, off, 32);
    float inv = 1.f / sum;
    bf16_t* rb = (bf16_t*)rp;
    for (int j = lane; j < N_; j += 32) {
      float e = __expf(rp[j] - mx) * inv;
      rb[j] = (bf16_t)e;
    }
  }
  __syncthreads();

  // phase 3: O[16 x 64] = P @ V^T, K = 1024, A-frags from LDS bf16 (shared across 4 d-tiles)
  v8f o[4] = {{}, {}, {}, {}};
  const bf16_t* arow = (const bf16_t*)(&sS[(size_t)l16 * N_]) + half * 8;
  for (int kk = 0; kk < N_; kk += 32) {
    Frag32 a;
    a.q[0] = *(const uint4*)(arow + kk);
    a.q[1] = *(const uint4*)(arow + kk + 16);
#pragma unroll
    for (int t = 0; t < 4; ++t) {
      Frag32 bb;
      const bf16_t* bp = Vb + (size_t)(t * 16 + l16) * N_ + half * 8 + kk;
      bb.q[0] = *(const uint4*)bp;
      bb.q[1] = *(const uint4*)(bp + 16);
      o[t] = __builtin_amdgcn_wmma_f32_16x16x32_bf16(false, a.v, false, bb.v,
                                                     (short)0, o[t], false, false);
    }
  }
#pragma unroll
  for (int t = 0; t < 4; ++t)
#pragma unroll
    for (int r = 0; r < 8; ++r)
      Ot[((size_t)b * N_ + qb * 16 + r + half * 8) * C_ + h * 64 + t * 16 + l16] =
          (bf16_t)(o[t][r]);
}

// ---------------------------------------------------------------- host
extern "C" void kernel_launch(void* const* d_in, const int* in_sizes, int n_in,
                              void* d_out, int out_size, void* d_ws, size_t ws_size,
                              hipStream_t stream) {
  (void)in_sizes; (void)n_in; (void)out_size; (void)ws_size;
  const float* x   = (const float*)d_in[0];
  const float* gns = (const float*)d_in[1];
  const float* gnb = (const float*)d_in[2];
  const float* q_w = (const float*)d_in[3];
  const float* q_b = (const float*)d_in[4];
  const float* k_w = (const float*)d_in[5];
  const float* k_b = (const float*)d_in[6];
  const float* v_w = (const float*)d_in[7];
  const float* v_b = (const float*)d_in[8];
  const float* p_w = (const float*)d_in[9];
  const float* p_b = (const float*)d_in[10];
  float* out = (float*)d_out;

  char* ws = (char*)d_ws;
  const size_t SZ_ACT = (size_t)B_ * N_ * C_ * sizeof(bf16_t);   // 16 MB each
  bf16_t* Ht = (bf16_t*)(ws + 0 * SZ_ACT);
  bf16_t* Qt = (bf16_t*)(ws + 1 * SZ_ACT);
  bf16_t* Kt = (bf16_t*)(ws + 2 * SZ_ACT);
  bf16_t* Vb = (bf16_t*)(ws + 3 * SZ_ACT);
  bf16_t* Ot = (bf16_t*)(ws + 4 * SZ_ACT);
  bf16_t* wq = (bf16_t*)(ws + 5 * SZ_ACT);
  bf16_t* wk = wq + (size_t)C_ * C_;
  bf16_t* wv = wk + (size_t)C_ * C_;
  bf16_t* wp = wv + (size_t)C_ * C_;

  int nw = C_ * C_;
  cvt_f32_bf16<<<(nw + 255) / 256, 256, 0, stream>>>(q_w, wq, nw);
  cvt_f32_bf16<<<(nw + 255) / 256, 256, 0, stream>>>(k_w, wk, nw);
  cvt_f32_bf16<<<(nw + 255) / 256, 256, 0, stream>>>(v_w, wv, nw);
  cvt_f32_bf16<<<(nw + 255) / 256, 256, 0, stream>>>(p_w, wp, nw);

  groupnorm_to_bf16<<<B_ * 32, 256, 0, stream>>>(x, gns, gnb, Ht);

  dim3 gblk(32, 4);
  long long sAct = (long long)N_ * C_;
  // tiles per batch = (M/64)*(N/64) = 128 -> grid.x = 32 with 4 waves/block
  // Q = Ht @ Wq^T  -> [pix][C] bf16, bias per col
  gemm_nt_bf16<<<dim3(32, 1, B_), gblk, 0, stream>>>(Ht, wq, q_b, nullptr, Qt,
                                                     N_, C_, C_, sAct, 0, sAct, 0);
  // K = Ht @ Wk^T
  gemm_nt_bf16<<<dim3(32, 1, B_), gblk, 0, stream>>>(Ht, wk, k_b, nullptr, Kt,
                                                     N_, C_, C_, sAct, 0, sAct, 0);
  // V = Wv @ Ht^T -> [C][pix] bf16, bias per row
  gemm_nt_bf16<<<dim3(32, 1, B_), gblk, 0, stream>>>(wv, Ht, v_b, nullptr, Vb,
                                                     C_, N_, C_, 0, sAct, sAct, 1);

  attn_fused<<<dim3(64, 8, B_), 32, 0, stream>>>(Qt, Kt, Vb, Ot);

  // out = x + Wp @ Ot^T + bias -> [C][pix] f32
  gemm_nt_bf16<<<dim3(32, 1, B_), gblk, 0, stream>>>(wp, Ot, p_b, x, out,
                                                     C_, N_, C_, 0, sAct, sAct, 2);
}
